// Attension_49503793054431
// MI455X (gfx1250) — matrix operand order repaired and checked by reading, not compiled, not run on hardware
//
#include <hip/hip_runtime.h>

#define SEQ    8192
#define DM     64
#define HD     16
#define WAVES  8                      // waves per block
#define NBLK   (SEQ / (16 * WAVES))   // 64 blocks, 512 query tiles total

typedef _Float16 v16h __attribute__((ext_vector_type(16)));
typedef _Float16 v8h  __attribute__((ext_vector_type(8)));
typedef float    v8f  __attribute__((ext_vector_type(8)));

// Q pre-scale: 1/sqrt(HD) * log2(e), so softmax uses raw v_exp_f32 (base-2).
#define QSCALE 0.36067376022224085f

// ---------------------------------------------------------------------------
// Kernel 1: QKV projection (fp32 math, f16 output).
//   Qh : plain   [row][16]
//   Kp : padded  [row][32] = [d0..7, 0 x8, d8..15, 0 x8]  (WMMA A-ready)
//   Vt : plain   V^T [d][SEQ]                              (remainder path)
//   Vs : V^T [d][SEQ], each 32-key group stored in plain order
//        [k0..31]                 (WMMA A-ready)
// ---------------------------------------------------------------------------
__global__ __launch_bounds__(256) void proj_kernel(
    const float* __restrict__ x,
    const float* __restrict__ Wq, const float* __restrict__ bq,
    const float* __restrict__ Wk, const float* __restrict__ bk,
    const float* __restrict__ Wv, const float* __restrict__ bv,
    _Float16* __restrict__ Qh, _Float16* __restrict__ Kp,
    _Float16* __restrict__ Vt, _Float16* __restrict__ Vs)
{
    int idx = blockIdx.x * blockDim.x + threadIdx.x;   // SEQ*HD threads
    if (idx >= SEQ * HD) return;
    int row = idx >> 4;
    int j   = idx & 15;
    const float* xr = x  + row * DM;
    const float* wq = Wq + j * DM;
    const float* wk = Wk + j * DM;
    const float* wv = Wv + j * DM;
    float aq = 0.f, ak = 0.f, av = 0.f;
    #pragma unroll
    for (int k = 0; k < DM; ++k) {
        float xv = xr[k];
        aq = fmaf(xv, wq[k], aq);
        ak = fmaf(xv, wk[k], ak);
        av = fmaf(xv, wv[k], av);
    }
    Qh[row * HD + j] = (_Float16)((aq + bq[j]) * QSCALE);

    const _Float16 hk = (_Float16)(ak + bk[j]);
    const int jpos = j + (j & 8);                      // 0..7 -> 0..7, 8..15 -> 16..23
    Kp[(size_t)row * 32 + jpos]     = hk;
    Kp[(size_t)row * 32 + jpos + 8] = (_Float16)0.0f;  // depth padding

    const _Float16 hv = (_Float16)(av + bv[j]);
    Vt[(size_t)j * SEQ + row] = hv;
    const int g   = row & 31;
    const int pos = g;
    Vs[(size_t)j * SEQ + (row & ~31) + pos] = hv;
}

// ---------------------------------------------------------------------------
// Kernel 2: causal flash attention, fully TRANSPOSED formulation:
//   S^T = K * Q^T   (A = Kp tile, B = Q^T)   -> lane owns ONE query row
//   O^T = V^T * P^T (A = Vs tile, B = P^T)   -> acc holds 8 contiguous dims
// All bulk A operands are single contiguous v16h loads (no register padding).
// ---------------------------------------------------------------------------
__global__ __launch_bounds__(WAVES * 32) void fa_kernel(
    const _Float16* __restrict__ Qh,
    const _Float16* __restrict__ Kp,
    const _Float16* __restrict__ Vt,
    const _Float16* __restrict__ Vs,
    float* __restrict__ out)
{
    __shared__ __align__(32) _Float16 ptile[WAVES][16 * 64];

    const int lane  = threadIdx.x & 31;
    const int wid   = threadIdx.x >> 5;
    const int tile  = wid * NBLK + blockIdx.x;        // 0..511 (interleaved)
    const int qbase = tile * 16;
    const int hi    = lane >> 4;                      // 0: lanes 0-15, 1: 16-31
    const int ln16  = lane & 15;

    const v16h zero16 = {};
    const v8h  zero8  = {};
    union AV { v16h v; v8h h[2]; };
    union P8 { v8h v; _Float16 e[8]; };

    // B operand of score WMMAs: Q^T (32x16); depth 16..31 zero (elements 8..15 of every lane).
    AV qa;
    qa.h[0] = *(const v8h*)(Qh + (size_t)(qbase + ln16) * HD + hi * 8);
    qa.h[1] = zero8;
    v16h qop = qa.v;

    float m_run = -3.0e38f;                           // per-lane row state
    float l_run = 0.0f;
    v8f acc = {};                                     // O^T: d = r + 8*hi

    _Float16* pt = ptile[wid];                        // P row-major [m][64]

    // ---------------- bulk: fully-unmasked 64-key chunks --------------------
    const int nbulk = tile >> 2;                      // keys [0, 64*nbulk)
    for (int kt = 0; kt < nbulk; ++kt) {
        const int kbase = kt * 64;

        // A operands: 4 K sub-tiles, each one contiguous 32B v16h load.
        const _Float16* kp = Kp + (size_t)(kbase + ln16) * 32 + hi * 16;
        v16h ka0 = *(const v16h*)(kp);
        v16h ka1 = *(const v16h*)(kp + 16 * 32);
        v16h ka2 = *(const v16h*)(kp + 32 * 32);
        v16h ka3 = *(const v16h*)(kp + 48 * 32);

        v8f c0 = {}, c1 = {}, c2 = {}, c3 = {};
        c0 = __builtin_amdgcn_wmma_f32_16x16x32_f16(
                 false, ka0, false, qop, (short)0, c0, false, false);
        c1 = __builtin_amdgcn_wmma_f32_16x16x32_f16(
                 false, ka1, false, qop, (short)0, c1, false, false);
        c2 = __builtin_amdgcn_wmma_f32_16x16x32_f16(
                 false, ka2, false, qop, (short)0, c2, false, false);
        c3 = __builtin_amdgcn_wmma_f32_16x16x32_f16(
                 false, ka3, false, qop, (short)0, c3, false, false);

        // ---- in-lane row max over 32 scores + one half-exchange -----------
        float t = c0[0];
        #pragma unroll
        for (int r = 0; r < 8; ++r) {
            t = fmaxf(t, c0[r]); t = fmaxf(t, c1[r]);
            t = fmaxf(t, c2[r]); t = fmaxf(t, c3[r]);
        }
        t = fmaxf(t, __shfl_xor(t, 16, 32));
        const float mnew = fmaxf(m_run, t);
        const float corr = exp2f(m_run - mnew);
        m_run = mnew;

        // ---- exponentials, in-lane sum, pack P rows ------------------------
        P8 q0, q1, q2, q3;
        float ps = 0.f;
        #pragma unroll
        for (int r = 0; r < 8; ++r) {
            float p0 = exp2f(c0[r] - mnew); ps += p0; q0.e[r] = (_Float16)p0;
            float p1 = exp2f(c1[r] - mnew); ps += p1; q1.e[r] = (_Float16)p1;
            float p2 = exp2f(c2[r] - mnew); ps += p2; q2.e[r] = (_Float16)p2;
            float p3 = exp2f(c3[r] - mnew); ps += p3; q3.e[r] = (_Float16)p3;
        }
        ps += __shfl_xor(ps, 16, 32);
        l_run = l_run * corr + ps;
        #pragma unroll
        for (int r = 0; r < 8; ++r) acc[r] *= corr;

        // ---- P to LDS: one b128 store per sub-tile -------------------------
        _Float16* prow = pt + ln16 * 64 + hi * 8;
        *(v8h*)(prow)      = q0.v;
        *(v8h*)(prow + 16) = q1.v;
        *(v8h*)(prow + 32) = q2.v;
        *(v8h*)(prow + 48) = q3.v;
        __builtin_amdgcn_wave_barrier();

        // ---- B operands: P^T tiles (32 keys each), contiguous 32B ----------
        v16h pb0 = *(const v16h*)(pt + ln16 * 64 + hi * 16);
        v16h pb1 = *(const v16h*)(pt + ln16 * 64 + 32 + hi * 16);
        __builtin_amdgcn_wave_barrier();

        // ---- A operands: plain V^T tiles, one 32B load each -------------
        const _Float16* vp = Vs + (size_t)ln16 * SEQ + kbase + hi * 16;
        v16h va0 = *(const v16h*)(vp);
        v16h va1 = *(const v16h*)(vp + 32);

        acc = __builtin_amdgcn_wmma_f32_16x16x32_f16(
                  false, va0, false, pb0, (short)0, acc, false, false);
        acc = __builtin_amdgcn_wmma_f32_16x16x32_f16(
                  false, va1, false, pb1, (short)0, acc, false, false);
    }

    // ---------------- remainder: masked 16-key diagonal tiles ---------------
    const int qrow = qbase + ln16;                    // this lane's query row
    for (int kbase = nbulk * 64; kbase < qbase + 16; kbase += 16) {
        // Kp rows are pre-padded: single contiguous load works here too.
        v16h ka = *(const v16h*)(Kp + (size_t)(kbase + ln16) * 32 + hi * 16);

        v8f c = {};
        c = __builtin_amdgcn_wmma_f32_16x16x32_f16(
                false, ka, false, qop, (short)0, c, false, false);

        float s[8];
        float t = -3.0e38f;
        #pragma unroll
        for (int r = 0; r < 8; ++r) {
            const int key = kbase + hi * 8 + r;
            s[r] = (key <= qrow) ? c[r] : -1.0e30f;   // causal mask
            t = fmaxf(t, s[r]);
        }
        t = fmaxf(t, __shfl_xor(t, 16, 32));
        const float mnew = fmaxf(m_run, t);
        const float corr = exp2f(m_run - mnew);
        m_run = mnew;

        P8 q0;
        float ps = 0.f;
        #pragma unroll
        for (int r = 0; r < 8; ++r) {
            float p = exp2f(s[r] - mnew); ps += p; q0.e[r] = (_Float16)p;
        }
        ps += __shfl_xor(ps, 16, 32);
        l_run = l_run * corr + ps;
        #pragma unroll
        for (int r = 0; r < 8; ++r) acc[r] *= corr;

        _Float16* prow = pt + ln16 * 64;
        *(v8h*)(prow + hi * 8)      = q0.v;           // keys 0..15: real P
        *(v8h*)(prow + 16 + hi * 8) = zero8;          // keys 16..31: zero pad
        __builtin_amdgcn_wave_barrier();

        v16h pb0 = *(const v16h*)(prow + hi * 16);
        __builtin_amdgcn_wave_barrier();

        AV va;                                        // plain V^T, k 16..31 = 0
        va.v = hi ? zero16 : *(const v16h*)(Vt + (size_t)ln16 * SEQ + kbase);

        acc = __builtin_amdgcn_wmma_f32_16x16x32_f16(
                  false, va.v, false, pb0, (short)0, acc, false, false);
    }

    // ---------------- epilogue: one contiguous 32B store per lane -----------
    const float rl = 1.0f / l_run;
    v8f res;
    #pragma unroll
    for (int r = 0; r < 8; ++r) res[r] = acc[r] * rl;
    *(v8f*)(out + (size_t)qrow * HD + hi * 8) = res;
}

// ---------------------------------------------------------------------------
extern "C" void kernel_launch(void* const* d_in, const int* in_sizes, int n_in,
                              void* d_out, int out_size, void* d_ws, size_t ws_size,
                              hipStream_t stream) {
    const float* x  = (const float*)d_in[0];
    const float* Wq = (const float*)d_in[1];
    const float* bq = (const float*)d_in[2];
    const float* Wk = (const float*)d_in[3];
    const float* bk = (const float*)d_in[4];
    const float* Wv = (const float*)d_in[5];
    const float* bv = (const float*)d_in[6];

    _Float16* Qh = (_Float16*)d_ws;            // 256 KB
    _Float16* Kp = Qh + (size_t)SEQ * HD;      // 512 KB (padded to depth 32)
    _Float16* Vt = Kp + (size_t)SEQ * 32;      // 256 KB (plain V^T)
    _Float16* Vs = Vt + (size_t)SEQ * HD;      // 256 KB (V^T, same layout as Vt)

    proj_kernel<<<(SEQ * HD) / 256, 256, 0, stream>>>(
        x, Wq, bq, Wk, bk, Wv, bv, Qh, Kp, Vt, Vs);

    fa_kernel<<<NBLK, WAVES * 32, 0, stream>>>(Qh, Kp, Vt, Vs, (float*)d_out);
}
